// GatConv_48524540510797
// MI455X (gfx1250) — matrix operand order repaired and checked by reading, not compiled, not run on hardware
//
#include <hip/hip_runtime.h>
#include <hip/hip_bf16.h>
#include <math.h>

#define IN_CH   256
#define NOUT    128   // HEADS * OUT_CH
#define HEADS   4
#define OC      32
#define NEGS    0.2f
#define EPSV    1e-16f

typedef __attribute__((ext_vector_type(16))) _Float16 v16h;
typedef __attribute__((ext_vector_type(8)))  float    v8f;

// ---------------------------------------------------------------------------
// Repack weight [256 x 128] f32 row-major -> per-(ktile,ntile,lane) contiguous
// f16 B-fragments for v_wmma_f32_16x16x32_f16.
// B frag layout (32x16, f16): lanes 0-15 hold K={0..7,16..23}, lanes 16-31 hold
// K={8..15,24..31}; N = lane & 15. Element i -> K = (lane>>4)*8 + (i&7) + (i>>3)*16.
// wpack index: (((kt*8 + nt)*32 + lane)*16 + i)
// ---------------------------------------------------------------------------
__global__ void gat_repack_w(const float* __restrict__ w, _Float16* __restrict__ wpack) {
    int gid = blockIdx.x * blockDim.x + threadIdx.x;
    if (gid >= 8 * 8 * 32 * 16) return;
    int i    = gid & 15;
    int lane = (gid >> 4) & 31;
    int nt   = (gid >> 9) & 7;
    int kt   = gid >> 12;
    int K  = kt * 32 + (lane >> 4) * 8 + (i & 7) + (i >> 3) * 16;
    int Nc = nt * 16 + (lane & 15);
    wpack[gid] = (_Float16)w[K * NOUT + Nc];
}

// ---------------------------------------------------------------------------
// out <- broadcast(bias); maxv <- -inf; sumv <- 0
// ---------------------------------------------------------------------------
__global__ void gat_init(float* __restrict__ out, float* __restrict__ maxv,
                         float* __restrict__ sumv, const float* __restrict__ bias,
                         int n_nodes) {
    int gid = blockIdx.x * blockDim.x + threadIdx.x;
    if (gid < n_nodes * NOUT) out[gid] = bias[gid & (NOUT - 1)];
    if (gid < n_nodes * HEADS) {
        maxv[gid] = -__builtin_inff();
        sumv[gid] = 0.0f;
    }
}

// ---------------------------------------------------------------------------
// h[M,128] = nodes[M,256] @ W[256,128]   (f16 WMMA, f32 accumulate)
// Block = 256 threads = 8 wave32s; each wave computes a 16x128 strip.
// B fragments are staged through LDS (double-buffered, 8KB per K-tile) so the
// 64KB packed weight is fetched from global once per block instead of once
// per wave; WMMAs then feed from ds_load while the next tile's global loads
// are in flight.
// A frag (16x32 f16): lanes 0-15 = rows, K {0..7,16..23}; lanes 16-31 same
// rows, K {8..15,24..31}.  (ISA 7.12.2 16-bit A layout)
// ---------------------------------------------------------------------------
__global__ __launch_bounds__(256)
void gat_gemm_wmma(const float* __restrict__ nodes, const _Float16* __restrict__ wpack,
                   float* __restrict__ h, int n_nodes) {
    __shared__ _Float16 bsh[2][8 * 32 * 16];   // 2 x 8KB B-tile buffers

    const int tid  = threadIdx.x;
    const int lane = tid & 31;
    const int wave = tid >> 5;
    const int r    = lane & 15;
    const int kh   = lane >> 4;           // which K-half this lane feeds
    const int mbase = blockIdx.x * 128 + wave * 16;
    const int mrow  = mbase + r;
    const int rowc  = mrow < n_nodes ? mrow : (n_nodes - 1);
    const float* __restrict__ arow = nodes + (size_t)rowc * IN_CH;

    // Cooperative stage of one K-tile of B frags: 4096 f16 = 8KB; 256 threads
    // move one v16h (32B) each.
    auto stage = [&](int kt, int buf) {
        const v16h* srcp = (const v16h*)(wpack + kt * 4096);
        ((v16h*)bsh[buf])[tid] = srcp[tid];
    };

    v8f acc[8] = {};

    stage(0, 0);
    __syncthreads();

    for (int kt = 0; kt < 8; ++kt) {
        const int buf = kt & 1;
        if (kt < 7) stage(kt + 1, buf ^ 1);   // overlap next tile with compute

        const int k0 = kt * 32;
        const float* ap = arow + k0 + kh * 8;
        // K offsets per lane: {0..7} and {16..23} shifted by kh*8
        float4 p0 = *(const float4*)(ap + 0);
        float4 p1 = *(const float4*)(ap + 4);
        float4 p2 = *(const float4*)(ap + 16);
        float4 p3 = *(const float4*)(ap + 20);
        if (kt < 7) __builtin_prefetch((const void*)(ap + 32), 0, 3); // global_prefetch_b8

        v16h a;
        a[0]  = (_Float16)p0.x; a[1]  = (_Float16)p0.y;
        a[2]  = (_Float16)p0.z; a[3]  = (_Float16)p0.w;
        a[4]  = (_Float16)p1.x; a[5]  = (_Float16)p1.y;
        a[6]  = (_Float16)p1.z; a[7]  = (_Float16)p1.w;
        a[8]  = (_Float16)p2.x; a[9]  = (_Float16)p2.y;
        a[10] = (_Float16)p2.z; a[11] = (_Float16)p2.w;
        a[12] = (_Float16)p3.x; a[13] = (_Float16)p3.y;
        a[14] = (_Float16)p3.z; a[15] = (_Float16)p3.w;

        #pragma unroll
        for (int nt = 0; nt < 8; ++nt) {
            const v16h b = *(const v16h*)(&bsh[buf][(nt * 32 + lane) << 4]);
            acc[nt] = __builtin_amdgcn_wmma_f32_16x16x32_f16(
                false, a, false, b, (short)0, acc[nt], false, false);
        }
        __syncthreads();
    }

    // C/D layout: VGPR j -> M = j + (lane>=16 ? 8 : 0); N = lane & 15
    const int rhi  = (lane >> 4) * 8;
    const int ncol = lane & 15;
    #pragma unroll
    for (int nt = 0; nt < 8; ++nt) {
        #pragma unroll
        for (int j = 0; j < 8; ++j) {
            int row = mbase + rhi + j;
            if (row < n_nodes) h[(size_t)row * NOUT + nt * 16 + ncol] = acc[nt][j];
        }
    }
}

// ---------------------------------------------------------------------------
// Per-node attention factors: ai[n,hd] = h[n,hd,:].att_i, aj likewise.
// ---------------------------------------------------------------------------
__global__ void gat_node_att(const float* __restrict__ h, const float* __restrict__ att,
                             float* __restrict__ ai, float* __restrict__ aj, int n_nodes) {
    int gid = blockIdx.x * blockDim.x + threadIdx.x;
    if (gid >= n_nodes * HEADS) return;
    int hd = gid & (HEADS - 1);
    int n  = gid >> 2;
    const float* hp = h + (size_t)n * NOUT + hd * OC;
    const float* ti = att + hd * (2 * OC);
    const float* tj = ti + OC;
    float si = 0.0f, sj = 0.0f;
    #pragma unroll
    for (int c = 0; c < OC; ++c) {
        float v = hp[c];
        si += v * ti[c];
        sj += v * tj[c];
    }
    ai[gid] = si;
    aj[gid] = sj;
}

// ---------------------------------------------------------------------------
// Ordered float atomic-max via signed/unsigned int trick (works across signs
// when initialized to -inf).
// ---------------------------------------------------------------------------
__device__ __forceinline__ void atomicMaxF(float* addr, float v) {
    if (v >= 0.0f) atomicMax((int*)addr, __float_as_int(v));
    else           atomicMin((unsigned int*)addr, __float_as_uint(v));
}

// alpha = leaky_relu(ai[dst]+aj[src]); segment max over dst
__global__ void gat_alpha(const int* __restrict__ adj, const float* __restrict__ ai,
                          const float* __restrict__ aj, float* __restrict__ ebuf,
                          float* __restrict__ maxv, int n_edges) {
    int e = blockIdx.x * blockDim.x + threadIdx.x;
    if (e >= n_edges) return;
    int dst = adj[e];
    int src = adj[n_edges + e];
    #pragma unroll
    for (int hd = 0; hd < HEADS; ++hd) {
        float al = ai[dst * HEADS + hd] + aj[src * HEADS + hd];
        al = al > 0.0f ? al : NEGS * al;
        ebuf[e * HEADS + hd] = al;
        atomicMaxF(&maxv[dst * HEADS + hd], al);
    }
}

// e = exp(alpha - m[dst]); segment sum over dst
__global__ void gat_expsum(const int* __restrict__ adj, float* __restrict__ ebuf,
                           const float* __restrict__ maxv, float* __restrict__ sumv,
                           int n_edges) {
    int e = blockIdx.x * blockDim.x + threadIdx.x;
    if (e >= n_edges) return;
    int dst = adj[e];
    #pragma unroll
    for (int hd = 0; hd < HEADS; ++hd) {
        // dst has at least this edge, so maxv[dst] is finite (matches the
        // reference's isfinite() masking which only affects edge-less nodes).
        float ex = __expf(ebuf[e * HEADS + hd] - maxv[dst * HEADS + hd]);
        ebuf[e * HEADS + hd] = ex;
        atomicAdd(&sumv[dst * HEADS + hd], ex);
    }
}

// out[dst, ch] += (e / (s[dst]+eps)) * h[src, ch]   -- 1 thread per (edge, ch)
__global__ void gat_scatter(const int* __restrict__ adj, const float* __restrict__ h,
                            const float* __restrict__ ebuf, const float* __restrict__ sumv,
                            float* __restrict__ out, int n_edges) {
    int gid = blockIdx.x * blockDim.x + threadIdx.x;
    if (gid >= n_edges * NOUT) return;
    int e  = gid >> 7;
    int ch = gid & (NOUT - 1);
    int hd = ch >> 5;
    int dst = adj[e];
    int src = adj[n_edges + e];
    float a = ebuf[e * HEADS + hd] / (sumv[dst * HEADS + hd] + EPSV);
    atomicAdd(out + (size_t)dst * NOUT + ch, a * h[(size_t)src * NOUT + ch]);
}

// ---------------------------------------------------------------------------
extern "C" void kernel_launch(void* const* d_in, const int* in_sizes, int n_in,
                              void* d_out, int out_size, void* d_ws, size_t ws_size,
                              hipStream_t stream) {
    const float* nodes  = (const float*)d_in[0];
    const int*   adj    = (const int*)d_in[1];
    const float* weight = (const float*)d_in[2];
    const float* att    = (const float*)d_in[3];
    const float* bias   = (const float*)d_in[4];
    float*       out    = (float*)d_out;

    const int n_nodes = in_sizes[0] / IN_CH;
    const int n_edges = in_sizes[1] / 2;

    char*  ws  = (char*)d_ws;
    size_t off = 0;
    auto alloc = [&](size_t bytes) -> void* {
        void* p = ws + off;
        off = (off + bytes + 255) & ~(size_t)255;
        return p;
    };
    float*    h     = (float*)alloc((size_t)n_nodes * NOUT * sizeof(float));
    _Float16* wpack = (_Float16*)alloc((size_t)8 * 8 * 32 * 16 * sizeof(_Float16));
    float*    ai    = (float*)alloc((size_t)n_nodes * HEADS * sizeof(float));
    float*    aj    = (float*)alloc((size_t)n_nodes * HEADS * sizeof(float));
    float*    maxv  = (float*)alloc((size_t)n_nodes * HEADS * sizeof(float));
    float*    sumv  = (float*)alloc((size_t)n_nodes * HEADS * sizeof(float));
    float*    ebuf  = (float*)alloc((size_t)n_edges * HEADS * sizeof(float));

    const int B = 256;
    gat_repack_w<<<(8 * 8 * 32 * 16 + B - 1) / B, B, 0, stream>>>(weight, wpack);
    gat_init<<<((size_t)n_nodes * NOUT + B - 1) / B, B, 0, stream>>>(out, maxv, sumv, bias, n_nodes);
    gat_gemm_wmma<<<(n_nodes + 127) / 128, B, 0, stream>>>(nodes, wpack, h, n_nodes);
    gat_node_att<<<((size_t)n_nodes * HEADS + B - 1) / B, B, 0, stream>>>(h, att, ai, aj, n_nodes);
    gat_alpha<<<(n_edges + B - 1) / B, B, 0, stream>>>(adj, ai, aj, ebuf, maxv, n_edges);
    gat_expsum<<<(n_edges + B - 1) / B, B, 0, stream>>>(adj, ebuf, maxv, sumv, n_edges);
    gat_scatter<<<((size_t)n_edges * NOUT + B - 1) / B, B, 0, stream>>>(adj, h, ebuf, sumv, out, n_edges);
}
